// Vanilla_Linear_Attention_25924422598783
// MI455X (gfx1250) — matrix operand order repaired and checked by reading, not compiled, not run on hardware
//
#include <hip/hip_runtime.h>

// ---------------------------------------------------------------------------
// Linear attention, algebraically fused:
//   G_h   = xh^T xh                       (per b,h; reduction over N=8192)
//   dots  = Wk G Wv^T / N ; M = Wq^T dots ; P_h = M_h Wo_h^T
//   out   = x @ P + bo                    (32768 x 1024 x 1024 GEMM)
// Heavy math via v_wmma_f32_16x16x32_f16; GEMM B-tiles streamed by the
// Tensor Data Mover (tensor_load_to_lds + s_wait_tensorcnt), double-buffered.
// ---------------------------------------------------------------------------

typedef __attribute__((ext_vector_type(16))) _Float16 v16h;
typedef __attribute__((ext_vector_type(8)))  _Float16 v8h;
typedef __attribute__((ext_vector_type(8)))  float    v8f;
typedef __attribute__((ext_vector_type(4)))  unsigned int u32x4;
typedef __attribute__((ext_vector_type(8)))  int i32x8;
typedef __attribute__((ext_vector_type(4)))  int i32x4;

#define NSEQ  8192
#define DIMF  1024
#define DHEAD 64

// ---------------------------- workspace zeroing ----------------------------
__global__ void zero_f32_kernel(float* __restrict__ p, int n) {
  int i = blockIdx.x * blockDim.x + threadIdx.x;
  if (i < n) p[i] = 0.0f;
}

// ------------------------------ Gram kernel --------------------------------
// grid (64, 8), block 128 (4 waves). Block (bh, chunk) accumulates the
// 64x64 Gram matrix of xh[b,h, chunk*1024 .. +1024, :] via WMMA and
// atomicAdd-reduces into G[bh].
__global__ __launch_bounds__(128)
void gram_kernel(const float* __restrict__ x, float* __restrict__ G) {
  __shared__ _Float16 XsT[64 * 144];   // XsT[d][t] = xh[n0+t, d], stride 144 (288B)

  const int bh   = blockIdx.x;
  const int b    = bh >> 4, h = bh & 15;
  const int tid  = threadIdx.x;
  const int lane = tid & 31;
  const int w    = tid >> 5;          // wave id 0..3 -> e-column tile
  const int hs   = lane >> 4;         // half-select (K pattern)
  const int l15  = lane & 15;

  v8f acc[4] = {};                    // d-tiles 0..3 x (e-tile w)

  const size_t rowbase = ((size_t)b * NSEQ) * DIMF + (size_t)h * DHEAD;

  for (int s = 0; s < 8; ++s) {
    const int n0 = blockIdx.y * 1024 + s * 128;
    __syncthreads();
    // load + transpose: 128 rows x 64 head-dims, fp32 -> f16
    {
      const float* src = x + rowbase + (size_t)(n0 + tid) * DIMF;
      #pragma unroll
      for (int d = 0; d < 64; d += 4) {
        float4 f = *(const float4*)(src + d);
        XsT[(d + 0) * 144 + tid] = (_Float16)f.x;
        XsT[(d + 1) * 144 + tid] = (_Float16)f.y;
        XsT[(d + 2) * 144 + tid] = (_Float16)f.z;
        XsT[(d + 3) * 144 + tid] = (_Float16)f.w;
      }
    }
    __syncthreads();

    #pragma unroll
    for (int kk = 0; kk < 128; kk += 32) {       // 4 k-steps of 32 over n
      // B fragment: column e = w*16 + l15, K = {kb..kb+7, kb+16..kb+23}, kb = hs*8
      v16h bf;
      {
        const _Float16* p = &XsT[(w * 16 + l15) * 144 + kk + hs * 8];
        v8h lo = *(const v8h*)p;
        v8h hi = *(const v8h*)(p + 16);
        #pragma unroll
        for (int i = 0; i < 8; ++i) { bf[i] = lo[i]; bf[i + 8] = hi[i]; }
      }
      #pragma unroll
      for (int dt = 0; dt < 4; ++dt) {
        // A fragment: row d = dt*16 + l15, K = {kb..kb+7, kb+16..kb+23}, kb = hs*8
        const _Float16* p = &XsT[(dt * 16 + l15) * 144 + kk + hs * 8];
        v8h lo = *(const v8h*)p;
        v8h hi = *(const v8h*)(p + 16);
        v16h af;
        #pragma unroll
        for (int i = 0; i < 8; ++i) { af[i] = lo[i]; af[i + 8] = hi[i]; }
        acc[dt] = __builtin_amdgcn_wmma_f32_16x16x32_f16(
            false, af, false, bf, (short)0, acc[dt], false, false);
      }
    }
  }

  // reduce into G[bh][64][64] (fp32)
  float* g = G + (size_t)bh * 4096;
  #pragma unroll
  for (int dt = 0; dt < 4; ++dt)
    #pragma unroll
    for (int v = 0; v < 8; ++v) {
      const int row = dt * 16 + v + hs * 8;
      atomicAdd(&g[row * 64 + (w * 16 + l15)], acc[dt][v]);
    }
}

// ------------------------------ fixup kernel -------------------------------
// grid 64, block 256. Tiny chained matmuls; writes P transposed (PbT[c][k], f16)
// so the big GEMM's B-tile loads are contiguous.
__global__ __launch_bounds__(256)
void fixup_kernel(const float* __restrict__ G,
                  const float* __restrict__ Wq, const float* __restrict__ Wk,
                  const float* __restrict__ Wv, const float* __restrict__ Wo,
                  _Float16* __restrict__ PbT) {
  __shared__ float sA[64 * 64];
  __shared__ float sB[64 * 64];
  const int bh = blockIdx.x, b = bh >> 4, h = bh & 15;
  const int tid = threadIdx.x;
  const float* g = G + (size_t)bh * 4096;
  const float invN = 1.0f / (float)NSEQ;

  // T1[a,e] = sum_bb G[a,bb] * Wv[e,bb]
  for (int idx = tid; idx < 4096; idx += 256) {
    const int a = idx >> 6, e = idx & 63;
    float s = 0.0f;
    #pragma unroll 8
    for (int bb = 0; bb < 64; ++bb) s += g[a * 64 + bb] * Wv[e * 64 + bb];
    sA[idx] = s;
  }
  __syncthreads();
  // dots[d,e] = invN * sum_a Wk[d,a] * T1[a,e]
  for (int idx = tid; idx < 4096; idx += 256) {
    const int d = idx >> 6, e = idx & 63;
    float s = 0.0f;
    #pragma unroll 8
    for (int a = 0; a < 64; ++a) s += Wk[d * 64 + a] * sA[a * 64 + e];
    sB[idx] = s * invN;
  }
  __syncthreads();
  // M[i,e] = sum_j Wq[j,i] * dots[j,e]
  for (int idx = tid; idx < 4096; idx += 256) {
    const int i = idx >> 6, e = idx & 63;
    float s = 0.0f;
    #pragma unroll 8
    for (int j = 0; j < 64; ++j) s += Wq[j * 64 + i] * sB[j * 64 + e];
    sA[idx] = s;            // sA now holds M (step-1 data fully consumed)
  }
  __syncthreads();
  // PbT[b][c][h*64+d] = sum_e M[d,e] * Wo[c, h*64+e]
  for (int idx = tid; idx < 65536; idx += 256) {
    const int c = idx >> 6, d = idx & 63;
    float s = 0.0f;
    #pragma unroll 8
    for (int e = 0; e < 64; ++e) s += sA[d * 64 + e] * Wo[(size_t)c * DIMF + h * 64 + e];
    PbT[(size_t)b * (DIMF * DIMF) + (size_t)c * DIMF + h * 64 + d] = (_Float16)s;
  }
}

// -------------------- TDM descriptor + issue (B tiles) ---------------------
// Moves a 128-row x 32-half tile of PbT into LDS, padding each 64B row with
// 32B so the LDS layout lands directly on the 48-half fragment stride.
// D# layout per CDNA5 ISA ch.8 (group0: count/lds/global/type, group1: dims).
__device__ __forceinline__ void issue_b_tile_tdm(const _Float16* gsrc,
                                                 unsigned lds_off) {
  const unsigned long long ga = (unsigned long long)gsrc;
  u32x4 g0;
  g0[0] = 1u;                                        // count=1, user desc
  g0[1] = lds_off;                                   // lds_addr (bytes)
  g0[2] = (unsigned)(ga & 0xFFFFFFFFu);              // global_addr[31:0]
  g0[3] = (unsigned)((ga >> 32) & 0x1FFFFFFu)        // global_addr[56:32]
        | (2u << 30);                                // type = 2 ("image")
  i32x8 g1;
  g1[0] = (1 << 16)      // data_size = 2 bytes (f16)
        | (1 << 20)      // pad_enable
        | (3 << 22)      // pad_interval: pad after 16 DWORDs (64B row)
        | (7 << 25);     // pad_amount: 8 DWORDs (32B) -> 96B row stride
  g1[1] = (int)(1024u << 16);   // tensor_dim0[15:0]=1024 (k length)
  g1[2] = (int)(1024u << 16);   // tensor_dim1[15:0]=1024 (c rows)
  g1[3] = (int)(32u << 16);     // tile_dim0 = 32 halves along k
  g1[4] = 128;                  // tile_dim1 = 128 rows; tile_dim2 = 0
  g1[5] = 1024;                 // tensor_dim0_stride = 1024 elements
  g1[6] = 0;
  g1[7] = 0;
  i32x4 z4 = {0, 0, 0, 0};
#if defined(__clang_major__) && (__clang_major__ >= 23)
  i32x8 z8 = {0, 0, 0, 0, 0, 0, 0, 0};
  __builtin_amdgcn_tensor_load_to_lds(g0, g1, z4, z4, z8, 0);
#else
  __builtin_amdgcn_tensor_load_to_lds(g0, g1, z4, z4, 0);
#endif
}

// ------------------------------- main GEMM ---------------------------------
// out[b] = x[b] (8192x1024, fp32) @ P[b] (1024x1024, f16) + bo, fp32 out.
// grid (8, 256), block 256 (8 waves). 128x128 block tile, K-tile 32.
// A tile: VALU fp32->f16 convert into LDS. B tile: TDM DMA, double-buffered.
__global__ __launch_bounds__(256)
void gemm_kernel(const float* __restrict__ x, const _Float16* __restrict__ PbT,
                 const float* __restrict__ bo, float* __restrict__ out) {
  __shared__ _Float16 As[128 * 48];        // As[m][k], stride 48 halves (96B)
  __shared__ _Float16 Bs[2][128 * 48];     // Bs[buf][n][k] = P[k][c0+n], TDM-filled

  const int tid    = threadIdx.x;
  const int lane   = tid & 31;
  const int wid    = tid >> 5;
  const int wave_m = wid >> 2;        // 0..1
  const int wave_n = wid & 3;         // 0..3
  const int hs     = lane >> 4;
  const int l15    = lane & 15;

  const int c0 = blockIdx.x * 128;
  const int r0 = blockIdx.y * 128;
  const int b  = r0 >> 13;                        // 8192 rows per batch
  const _Float16* Pb = PbT + (size_t)b * (DIMF * DIMF);

  const int lm = tid >> 1;            // 0..127: tile row handled by this thread
  const int lk = (tid & 1) * 16;      // 0 or 16: k-segment

  v8f acc[4][2] = {};

  // prologue: DMA first B tile into buffer 0
  if (wid == 0) {
    issue_b_tile_tdm(Pb + (size_t)c0 * DIMF, (unsigned)(size_t)&Bs[0][0]);
  }

  for (int k0 = 0; k0 < DIMF; k0 += 32) {
    const int cur = (k0 >> 5) & 1;
    // A tile: x[r0+lm][k0+lk..+15] fp32 -> f16 into LDS (overlaps B DMA)
    {
      const float* src = x + (size_t)(r0 + lm) * DIMF + k0 + lk;
      _Float16* dst = &As[lm * 48 + lk];
      #pragma unroll
      for (int i = 0; i < 4; ++i) {
        float4 f = *(const float4*)(src + i * 4);
        dst[i * 4 + 0] = (_Float16)f.x;
        dst[i * 4 + 1] = (_Float16)f.y;
        dst[i * 4 + 2] = (_Float16)f.z;
        dst[i * 4 + 3] = (_Float16)f.w;
      }
    }
    if (k0 + 32 < DIMF) {   // hint next A k-tile toward the caches
      __builtin_prefetch(x + (size_t)(r0 + lm) * DIMF + k0 + 32 + lk, 0, 0);
    }
    if (wid == 0) __builtin_amdgcn_s_wait_tensorcnt(0);   // B[cur] DMA done
    __syncthreads();                                      // publish As + Bs[cur]
    // kick DMA for the next k-tile into the other buffer while we compute
    if (wid == 0 && (k0 + 32) < DIMF) {
      issue_b_tile_tdm(Pb + (size_t)c0 * DIMF + (k0 + 32),
                       (unsigned)(size_t)&Bs[cur ^ 1][0]);
    }

    v16h a[4], bf[2];
    #pragma unroll
    for (int mt = 0; mt < 4; ++mt) {
      const int m = wave_m * 64 + mt * 16 + l15;
      const _Float16* p = &As[m * 48 + hs * 8];
      v8h lo = *(const v8h*)p;
      v8h hi = *(const v8h*)(p + 16);
      #pragma unroll
      for (int i = 0; i < 8; ++i) { a[mt][i] = lo[i]; a[mt][i + 8] = hi[i]; }
    }
    #pragma unroll
    for (int nt = 0; nt < 2; ++nt) {
      const int n = wave_n * 32 + nt * 16 + l15;
      const _Float16* p = &Bs[cur][n * 48 + hs * 8];
      v8h lo = *(const v8h*)p;
      v8h hi = *(const v8h*)(p + 16);
      #pragma unroll
      for (int i = 0; i < 8; ++i) { bf[nt][i] = lo[i]; bf[nt][i + 8] = hi[i]; }
    }
    #pragma unroll
    for (int mt = 0; mt < 4; ++mt)
      #pragma unroll
      for (int nt = 0; nt < 2; ++nt)
        acc[mt][nt] = __builtin_amdgcn_wmma_f32_16x16x32_f16(
            false, a[mt], false, bf[nt], (short)0, acc[mt][nt], false, false);

    __syncthreads();   // consumers done: As may be overwritten next iter
  }

  // epilogue: fp32 store + bias
  const int cA = c0 + wave_n * 32 + l15;
  const float bo0 = bo[cA];
  const float bo1 = bo[cA + 16];
  #pragma unroll
  for (int mt = 0; mt < 4; ++mt) {
    #pragma unroll
    for (int v = 0; v < 8; ++v) {
      const int row = r0 + wave_m * 64 + mt * 16 + v + hs * 8;
      float* o = out + (size_t)row * DIMF;
      o[cA]      = acc[mt][0][v] + bo0;
      o[cA + 16] = acc[mt][1][v] + bo1;
    }
  }
}

// ------------------------------- launcher ----------------------------------
extern "C" void kernel_launch(void* const* d_in, const int* in_sizes, int n_in,
                              void* d_out, int out_size, void* d_ws, size_t ws_size,
                              hipStream_t stream) {
  const float* x  = (const float*)d_in[0];
  const float* Wq = (const float*)d_in[1];
  const float* Wk = (const float*)d_in[2];
  const float* Wv = (const float*)d_in[3];
  const float* Wo = (const float*)d_in[4];
  const float* bo = (const float*)d_in[5];
  float* out = (float*)d_out;

  // workspace layout: G (64 * 64*64 fp32 = 1 MB) | PbT (4 * 1024*1024 f16 = 8 MB)
  float*    G   = (float*)d_ws;
  _Float16* PbT = (_Float16*)((char*)d_ws + (size_t)64 * 4096 * sizeof(float));

  zero_f32_kernel<<<1024, 256, 0, stream>>>(G, 64 * 4096);
  gram_kernel   <<<dim3(64, 8), 128, 0, stream>>>(x, G);
  fixup_kernel  <<<64, 256, 0, stream>>>(G, Wq, Wk, Wv, Wo, PbT);
  gemm_kernel   <<<dim3(8, 256), 256, 0, stream>>>(x, PbT, bo, out);
}
